// LDocumentDecoder_65524021067838
// MI455X (gfx1250) — matrix operand order repaired and checked by reading, not compiled, not run on hardware
//
#include <hip/hip_runtime.h>
#include <hip/hip_bf16.h>
#include <math.h>

#define PAD_TOK 1
#define SOD_TOK 2
#define VOCAB   32000
#define EMB     512
#define HID     512
#define BSZ     8
#define DOCLEN  16
#define SENTLEN 40
#define SRCLEN  400
#define SRCPAD  416               // SRCLEN padded to multiple of 32 for K-tiling
#define NB2     (BSZ*DOCLEN)      // 128
#define GATES   (4*HID)           // 2048
#define NEGV    (-1e9f)

typedef __bf16 bf16_t;
typedef bf16_t v16bf __attribute__((ext_vector_type(16)));
typedef bf16_t v8bf  __attribute__((ext_vector_type(8)));
typedef float  v8f   __attribute__((ext_vector_type(8)));

static __device__ __forceinline__ v8f wmma_bf16(v16bf a, v16bf b, v8f c) {
  // D = A(16x32 bf16) * B(32x16 bf16) + C(16x16 f32)
  return __builtin_amdgcn_wmma_f32_16x16x32_bf16(false, a, false, b, (short)0, c, false, false);
}

static __device__ __forceinline__ float sigm(float x) { return 1.0f / (1.0f + __expf(-x)); }

static __device__ __forceinline__ v8bf ld8(const bf16_t* p) { return *(const v8bf*)p; }

// ---------------- WMMA fragment loaders (CDNA5 ISA 7.12.2), vectorized ----------------
// A (16x32 bf16): per lane the 16 elements are two contiguous 8-element groups:
//   k in [k0+half*8, +8) and [k0+16+half*8, +8)  -> two 16-byte loads.
static __device__ __forceinline__ v16bf frag_a_bf16(const bf16_t* s, int lda, int k0, int lane) {
  const int half = lane >> 4, m = lane & 15;
  const bf16_t* base = s + m * lda + k0 + half * 8;
  v8bf g0 = ld8(base);
  v8bf g1 = ld8(base + 16);
  return __builtin_shufflevector(g0, g1, 0,1,2,3,4,5,6,7,8,9,10,11,12,13,14,15);
}
// A = concat[ctx(bf16,512) | h(bf16,512)], K = 1024; 8-groups never straddle the 512 boundary.
static __device__ __forceinline__ v16bf frag_a_cat(const bf16_t* ctx, const bf16_t* h, int k0, int lane) {
  const int half = lane >> 4, m = lane & 15;
  const int ka = k0 + half * 8, kb = ka + 16;
  v8bf g0 = (ka < HID) ? ld8(ctx + m * HID + ka) : ld8(h + m * HID + (ka - HID));
  v8bf g1 = (kb < HID) ? ld8(ctx + m * HID + kb) : ld8(h + m * HID + (kb - HID));
  return __builtin_shufflevector(g0, g1, 0,1,2,3,4,5,6,7,8,9,10,11,12,13,14,15);
}
// B (32x16) from weight stored [N][K] bf16: per lane two 8-element groups (same k order as A) -> two 16-byte loads.
static __device__ __forceinline__ v16bf frag_b_nk(const bf16_t* w, int ldk, int n0, int k0, int lane) {
  const int khalf = lane >> 4, n = n0 + (lane & 15);
  const bf16_t* base = w + (size_t)n * ldk + k0 + khalf * 8;
  v8bf g0 = ld8(base);
  v8bf g1 = ld8(base + 16);
  return __builtin_shufflevector(g0, g1, 0,1,2,3,4,5,6,7,8,9,10,11,12,13,14,15);
}

// ---------------- shared LSTM building blocks ----------------
// Gate buffer stored TRANSPOSED [n][m] so each lane's 8 D-values are one 16B store.
union GBuf { bf16_t gT[GATES * 16]; bf16_t qcb[16 * HID]; };

static __device__ void gates_wmma(const bf16_t* xb, const bf16_t* hb,
                                  const bf16_t* wih, const bf16_t* whh,
                                  const float* bih, const float* bhh, bf16_t* gT) {
  const int wave = threadIdx.x >> 5, lane = threadIdx.x & 31;
  const int mh = lane >> 4, nl = lane & 15;
  for (int tile = wave; tile < GATES / 16; tile += 8) {
    const int n0 = tile * 16, n = n0 + nl;
    const float bias = bih[n] + bhh[n];
    v8f acc;
#pragma unroll
    for (int r = 0; r < 8; ++r) acc[r] = bias;
    for (int k0 = 0; k0 < HID; k0 += 32)
      acc = wmma_bf16(frag_a_bf16(xb, HID, k0, lane), frag_b_nk(wih, HID, n0, k0, lane), acc);
    for (int k0 = 0; k0 < HID; k0 += 32)
      acc = wmma_bf16(frag_a_bf16(hb, HID, k0, lane), frag_b_nk(whh, HID, n0, k0, lane), acc);
    v8bf pk;
#pragma unroll
    for (int r = 0; r < 8; ++r) pk[r] = (bf16_t)acc[r];
    *(v8bf*)(gT + n * 16 + mh * 8) = pk;   // transposed, 16B store
  }
}

static __device__ void lstm_elem(const bf16_t* gT, bf16_t* h, bf16_t* c) {
  for (int e = threadIdx.x; e < 16 * HID; e += 256) {
    const int m = e >> 9, d = e & (HID - 1);
    const float gi = (float)gT[(d)           * 16 + m];
    const float gf = (float)gT[(HID + d)     * 16 + m];
    const float gg = (float)gT[(2 * HID + d) * 16 + m];
    const float go = (float)gT[(3 * HID + d) * 16 + m];
    const float cc = sigm(gf) * (float)c[e] + sigm(gi) * tanhf(gg);
    c[e] = (bf16_t)cc;
    h[e] = (bf16_t)(sigm(go) * tanhf(cc));
  }
}

// ---------------- prep kernels ----------------
__global__ void k_cvt(const float* __restrict__ s, bf16_t* __restrict__ d, int n) {
  for (int i = blockIdx.x * blockDim.x + threadIdx.x; i < n; i += gridDim.x * blockDim.x)
    d[i] = (bf16_t)s[i];
}
// enc1 [g][j][d] -> bf16 transposed [g][d][j] with j padded to 416 (zeros)
__global__ void k_tr(const float* __restrict__ s, bf16_t* __restrict__ d) {
  const int n = BSZ * HID * SRCPAD;
  for (int i = blockIdx.x * blockDim.x + threadIdx.x; i < n; i += gridDim.x * blockDim.x) {
    const int j = i % SRCPAD, dd = (i / SRCPAD) % HID, g = i / (SRCPAD * HID);
    d[i] = (j < SRCLEN) ? (bf16_t)s[((size_t)g * SRCLEN + j) * HID + dd] : (bf16_t)0.0f;
  }
}
__global__ void k_zero(float* p, int n) {
  for (int i = blockIdx.x * blockDim.x + threadIdx.x; i < n; i += gridDim.x * blockDim.x)
    p[i] = 0.0f;
}

// ---------------- document-level LSTM + attention (1 workgroup, 16 steps) ----------------
__global__ __launch_bounds__(256)
void k_doc(const float* __restrict__ enc0, const unsigned char* __restrict__ mask,
           const float* __restrict__ emb,
           const bf16_t* __restrict__ wih, const bf16_t* __restrict__ whh,
           const float* __restrict__ bih, const float* __restrict__ bhh,
           const bf16_t* __restrict__ wq, const bf16_t* __restrict__ wo,
           float* __restrict__ sent_states, float* __restrict__ last_out) {
  __shared__ __align__(16) bf16_t sh[16 * HID];
  __shared__ __align__(16) bf16_t scst[16 * HID];
  __shared__ __align__(16) bf16_t sx[16 * HID];
  __shared__ __align__(16) bf16_t sscb[16 * SRCPAD];
  __shared__ __align__(16) GBuf   U;

  const int tid = threadIdx.x, wave = tid >> 5, lane = tid & 31;
  const int mh = lane >> 4, nl = lane & 15;

  // h0 = c0 = mean(enc0, axis=srclen); x0 = emb[SOD]
  for (int e = tid; e < 16 * HID; e += 256) {
    const int m = e >> 9, d = e & 511;
    float v = 0.0f;
    if (m < BSZ) {
      float s = 0.0f;
      for (int j = 0; j < SRCLEN; ++j) s += enc0[(m * SRCLEN + j) * HID + d];
      v = s * (1.0f / (float)SRCLEN);
    }
    sh[e] = (bf16_t)v;
    scst[e] = (bf16_t)v;
    sx[e] = (bf16_t)emb[SOD_TOK * EMB + d];
  }
  __syncthreads();

  for (int t = 0; t < DOCLEN; ++t) {
    gates_wmma(sx, sh, wih, whh, bih, bhh, U.gT);
    __syncthreads();
    lstm_elem(U.gT, sh, scst);
    __syncthreads();

    // q = h @ Wq^T  (WMMA, stored bf16 in U.qcb)
    for (int tile = wave; tile < HID / 16; tile += 8) {
      const int n0 = tile * 16, n = n0 + nl;
      v8f acc;
#pragma unroll
      for (int r = 0; r < 8; ++r) acc[r] = 0.0f;
      for (int k0 = 0; k0 < HID; k0 += 32)
        acc = wmma_bf16(frag_a_bf16(sh, HID, k0, lane), frag_b_nk(wq, HID, n0, k0, lane), acc);
#pragma unroll
      for (int r = 0; r < 8; ++r) U.qcb[(mh * 8 + r) * HID + n] = (bf16_t)acc[r];
    }
    __syncthreads();

    // per-row attention (each batch row has its own encoder rows): scalar fp32
    for (int rr = 0; rr < 2; ++rr) {
      const int m = wave + rr * 8;
      const int b = m & 7;
      const float* eb = enc0 + (size_t)b * SRCLEN * HID;
      for (int j = lane; j < SRCLEN; j += 32) {
        const float4* er = (const float4*)(eb + j * HID);
        float s = 0.0f;
        for (int d4 = 0; d4 < HID / 4; ++d4) {
          const float4 ev = er[d4];
          s = fmaf(ev.x, (float)U.qcb[m * HID + d4 * 4 + 0], s);
          s = fmaf(ev.y, (float)U.qcb[m * HID + d4 * 4 + 1], s);
          s = fmaf(ev.z, (float)U.qcb[m * HID + d4 * 4 + 2], s);
          s = fmaf(ev.w, (float)U.qcb[m * HID + d4 * 4 + 3], s);
        }
        sscb[m * SRCPAD + j] = (bf16_t)(mask[b * SRCLEN + j] ? NEGV : s);
      }
      float mx = -INFINITY;
      for (int j = lane; j < SRCLEN; j += 32) mx = fmaxf(mx, (float)sscb[m * SRCPAD + j]);
      for (int o = 16; o > 0; o >>= 1) mx = fmaxf(mx, __shfl_xor(mx, o, 32));
      float sum = 0.0f;
      for (int j = lane; j < SRCLEN; j += 32) sum += __expf((float)sscb[m * SRCPAD + j] - mx);
      for (int o = 16; o > 0; o >>= 1) sum += __shfl_xor(sum, o, 32);
      const float inv = 1.0f / sum;
      for (int j = lane; j < SRCLEN; j += 32)
        sscb[m * SRCPAD + j] = (bf16_t)(__expf((float)sscb[m * SRCPAD + j] - mx) * inv);
      __syncthreads();  // uniform: rr loop identical across all waves
      // ctx[m][d] = sum_j p[m][j] * enc[b][j][d]  (overwrites q row m; q is dead)
      for (int d = lane; d < HID; d += 32) {
        float s = 0.0f;
        for (int j = 0; j < SRCLEN; ++j) s = fmaf((float)sscb[m * SRCPAD + j], eb[j * HID + d], s);
        U.qcb[m * HID + d] = (bf16_t)s;
      }
      __syncthreads();
    }

    // out = tanh(concat[ctx,h] @ Wo^T); feeds next x, sent_states, last_out
    for (int tile = wave; tile < HID / 16; tile += 8) {
      const int n0 = tile * 16, n = n0 + nl;
      v8f acc;
#pragma unroll
      for (int r = 0; r < 8; ++r) acc[r] = 0.0f;
      for (int k0 = 0; k0 < 2 * HID; k0 += 32)
        acc = wmma_bf16(frag_a_cat(U.qcb, sh, k0, lane), frag_b_nk(wo, 2 * HID, n0, k0, lane), acc);
#pragma unroll
      for (int r = 0; r < 8; ++r) {
        const int m = mh * 8 + r;
        const float o = tanhf(acc[r]);
        sx[m * HID + n] = (bf16_t)o;
        if (m < BSZ) {
          sent_states[(m * DOCLEN + t) * HID + n] = o;
          if (t == DOCLEN - 1) last_out[m * HID + n] = o;
        }
      }
    }
    __syncthreads();
  }
}

// ---------------- word-level LSTM + attention (8 WGs x 16 sentences, 40 steps) ----------------
__global__ __launch_bounds__(256)
void k_word(const int* __restrict__ tokens, const float* __restrict__ spe,
            const float* __restrict__ emb, const bf16_t* __restrict__ enc1b,
            const bf16_t* __restrict__ enc1T, const unsigned char* __restrict__ mask,
            const bf16_t* __restrict__ wih, const bf16_t* __restrict__ whh,
            const float* __restrict__ bih, const float* __restrict__ bhh,
            const bf16_t* __restrict__ wq, const bf16_t* __restrict__ wo,
            const float* __restrict__ sent_states, bf16_t* __restrict__ wouts,
            float* __restrict__ watt) {
  __shared__ __align__(16) bf16_t sh[16 * HID];
  __shared__ __align__(16) bf16_t scst[16 * HID];
  __shared__ __align__(16) bf16_t sx[16 * HID];
  __shared__ __align__(16) bf16_t ssp[16 * HID];
  __shared__ __align__(16) bf16_t sscb[16 * SRCPAD];
  __shared__ __align__(16) GBuf   U;
  __shared__ int spos[16];

  const int g = blockIdx.x;   // batch index; sentences b2 = g*16 + m all share encoder_out1[g]
  const int tid = threadIdx.x, wave = tid >> 5, lane = tid & 31;
  const int mh = lane >> 4, nl = lane & 15;
  const bf16_t* eb  = enc1b + (size_t)g * SRCLEN * HID;   // [j][d] for scores
  const bf16_t* ebT = enc1T + (size_t)g * HID * SRCPAD;   // [d][jpad] for ctx
  const unsigned char* mb = mask + g * SRCLEN;

  if (tid < 16) {
    int ne = 0;
    for (int s = 0; s < SENTLEN; ++s) ne |= (tokens[(g * DOCLEN + tid) * SENTLEN + s] != PAD_TOK);
    spos[tid] = ne ? (PAD_TOK + 1 + tid) : 1;
  }
  // zero the K-tail of the probability buffer once (never overwritten)
  if (tid < 256) {
    const int m = tid >> 4, j = SRCLEN + (tid & 15);
    sscb[m * SRCPAD + j] = (bf16_t)0.0f;
  }
  __syncthreads();
  for (int e = tid; e < 16 * HID; e += 256) {
    const int m = e >> 9, d = e & 511;
    ssp[e] = (bf16_t)spe[spos[m] * EMB + d];
    const float hv = sent_states[(g * DOCLEN + m) * HID + d];
    sh[e] = (bf16_t)hv;
    scst[e] = (bf16_t)hv;
  }
  __syncthreads();

  for (int t = 0; t < SENTLEN; ++t) {
    // x_t = emb[token] + sent_pos   (vectorized: 8 elems per iteration)
    for (int e = tid; e < 16 * (HID / 8); e += 256) {
      const int m = e >> 6, d8 = (e & 63) * 8;
      const int tok = tokens[(g * DOCLEN + m) * SENTLEN + t];
      const float4* ep = (const float4*)(emb + (size_t)tok * EMB + d8);
      const float4 e0 = ep[0], e1 = ep[1];
      const v8bf sp = ld8(ssp + m * HID + d8);
      v8bf xv;
      xv[0] = (bf16_t)(e0.x + (float)sp[0]); xv[1] = (bf16_t)(e0.y + (float)sp[1]);
      xv[2] = (bf16_t)(e0.z + (float)sp[2]); xv[3] = (bf16_t)(e0.w + (float)sp[3]);
      xv[4] = (bf16_t)(e1.x + (float)sp[4]); xv[5] = (bf16_t)(e1.y + (float)sp[5]);
      xv[6] = (bf16_t)(e1.z + (float)sp[6]); xv[7] = (bf16_t)(e1.w + (float)sp[7]);
      *(v8bf*)(sx + m * HID + d8) = xv;
    }
    __syncthreads();
    gates_wmma(sx, sh, wih, whh, bih, bhh, U.gT);
    __syncthreads();
    lstm_elem(U.gT, sh, scst);
    __syncthreads();

    // q = h @ Wq^T (bf16 out)
    for (int tile = wave; tile < HID / 16; tile += 8) {
      const int n0 = tile * 16, n = n0 + nl;
      v8f acc;
#pragma unroll
      for (int r = 0; r < 8; ++r) acc[r] = 0.0f;
      for (int k0 = 0; k0 < HID; k0 += 32)
        acc = wmma_bf16(frag_a_bf16(sh, HID, k0, lane), frag_b_nk(wq, HID, n0, k0, lane), acc);
#pragma unroll
      for (int r = 0; r < 8; ++r) U.qcb[(mh * 8 + r) * HID + n] = (bf16_t)acc[r];
    }
    __syncthreads();

    // scores[16][400] = q @ enc^T  (shared-encoder GEMM)
    for (int tile = wave; tile < SRCLEN / 16; tile += 8) {
      const int n0 = tile * 16, n = n0 + nl;
      v8f acc;
#pragma unroll
      for (int r = 0; r < 8; ++r) acc[r] = 0.0f;
      for (int k0 = 0; k0 < HID; k0 += 32)
        acc = wmma_bf16(frag_a_bf16(U.qcb, HID, k0, lane), frag_b_nk(eb, HID, n0, k0, lane), acc);
      const bool msk = (mb[n] != 0);
#pragma unroll
      for (int r = 0; r < 8; ++r)
        sscb[(mh * 8 + r) * SRCPAD + n] = (bf16_t)(msk ? NEGV : acc[r]);
    }
    __syncthreads();

    // softmax per row (wave handles rows {wave, wave+8}); emit word_atte_scores
    for (int rr = 0; rr < 2; ++rr) {
      const int m = wave + rr * 8;
      float mx = -INFINITY;
      for (int j = lane; j < SRCLEN; j += 32) mx = fmaxf(mx, (float)sscb[m * SRCPAD + j]);
      for (int o = 16; o > 0; o >>= 1) mx = fmaxf(mx, __shfl_xor(mx, o, 32));
      float sum = 0.0f;
      for (int j = lane; j < SRCLEN; j += 32) sum += __expf((float)sscb[m * SRCPAD + j] - mx);
      for (int o = 16; o > 0; o >>= 1) sum += __shfl_xor(sum, o, 32);
      const float inv = 1.0f / sum;
      for (int j = lane; j < SRCLEN; j += 32) {
        const float p = __expf((float)sscb[m * SRCPAD + j] - mx) * inv;
        sscb[m * SRCPAD + j] = (bf16_t)p;
        watt[((size_t)(g * DOCLEN + m) * SENTLEN + t) * SRCLEN + j] = p;
      }
    }
    __syncthreads();

    // ctx[16][512] = p @ enc   (K = 416 padded; B from transposed [d][jpad] -> contiguous)
    for (int tile = wave; tile < HID / 16; tile += 8) {
      const int n0 = tile * 16, n = n0 + nl;
      v8f acc;
#pragma unroll
      for (int r = 0; r < 8; ++r) acc[r] = 0.0f;
      for (int k0 = 0; k0 < SRCPAD; k0 += 32)
        acc = wmma_bf16(frag_a_bf16(sscb, SRCPAD, k0, lane),
                        frag_b_nk(ebT, SRCPAD, n0, k0, lane), acc);
#pragma unroll
      for (int r = 0; r < 8; ++r) U.qcb[(mh * 8 + r) * HID + n] = (bf16_t)acc[r];
    }
    __syncthreads();

    // out = tanh(concat[ctx,h] @ Wo^T) -> w_outs (bf16, feeds fc GEMM)
    for (int tile = wave; tile < HID / 16; tile += 8) {
      const int n0 = tile * 16, n = n0 + nl;
      v8f acc;
#pragma unroll
      for (int r = 0; r < 8; ++r) acc[r] = 0.0f;
      for (int k0 = 0; k0 < 2 * HID; k0 += 32)
        acc = wmma_bf16(frag_a_cat(U.qcb, sh, k0, lane), frag_b_nk(wo, 2 * HID, n0, k0, lane), acc);
#pragma unroll
      for (int r = 0; r < 8; ++r) {
        const int m = mh * 8 + r;
        wouts[((size_t)(g * DOCLEN + m) * SENTLEN + t) * HID + n] = (bf16_t)tanhf(acc[r]);
      }
    }
    __syncthreads();
  }
}

// ---------------- output projection: [5120,512]bf16 @ [32000,512]^T bf16 -> f32 ----------------
__global__ __launch_bounds__(256)
void k_fc(const bf16_t* __restrict__ A, const bf16_t* __restrict__ W,
          const float* __restrict__ bias, float* __restrict__ out) {
  const int tid = threadIdx.x, wave = tid >> 5, lane = tid & 31;
  const int mh = lane >> 4, nl = lane & 15;
  const int wm = wave >> 1, wn = wave & 1;             // 4 (M) x 2 (N) wave grid
  const int m0 = blockIdx.y * 64 + wm * 16;
  const int nbase = blockIdx.x * 128 + wn * 64;

  v8f acc[4];
#pragma unroll
  for (int i = 0; i < 4; ++i) {
    const float b = bias[nbase + i * 16 + nl];
#pragma unroll
    for (int r = 0; r < 8; ++r) acc[i][r] = b;
  }
  const bf16_t* Arow = A + (size_t)m0 * HID;
  for (int k0 = 0; k0 < HID; k0 += 32) {
    if (k0 + 32 < HID)
      __builtin_prefetch(&W[(size_t)(nbase + nl) * HID + k0 + 32], 0, 0);
    const v16bf af = frag_a_bf16(Arow, HID, k0, lane);
#pragma unroll
    for (int i = 0; i < 4; ++i)
      acc[i] = wmma_bf16(af, frag_b_nk(W, HID, nbase + i * 16, k0, lane), acc[i]);
  }
#pragma unroll
  for (int i = 0; i < 4; ++i) {
    const int n = nbase + i * 16 + nl;
#pragma unroll
    for (int r = 0; r < 8; ++r)
      out[(size_t)(m0 + mh * 8 + r) * VOCAB + n] = acc[i][r];
  }
}

// ---------------- host side ----------------
extern "C" void kernel_launch(void* const* d_in, const int* in_sizes, int n_in,
                              void* d_out, int out_size, void* d_ws, size_t ws_size,
                              hipStream_t stream) {
  const int*   tokens = (const int*)d_in[0];
  const float* enc0   = (const float*)d_in[1];
  const float* enc1   = (const float*)d_in[2];
  const unsigned char* mask = (const unsigned char*)d_in[3];
  const float* emb  = (const float*)d_in[4];
  const float* spe  = (const float*)d_in[5];
  const float* wihd = (const float*)d_in[6];
  const float* whhd = (const float*)d_in[7];
  const float* bihd = (const float*)d_in[8];
  const float* bhhd = (const float*)d_in[9];
  const float* aqd  = (const float*)d_in[10];
  const float* aod  = (const float*)d_in[11];
  const float* wihw = (const float*)d_in[12];
  const float* whhw = (const float*)d_in[13];
  const float* bihw = (const float*)d_in[14];
  const float* bhhw = (const float*)d_in[15];
  const float* aqw  = (const float*)d_in[16];
  const float* aow  = (const float*)d_in[17];
  const float* fcw  = (const float*)d_in[18];
  const float* fcb  = (const float*)d_in[19];
  float* out = (float*)d_out;

  // output regions (flat concat, element offsets)
  const size_t OFF1 = (size_t)BSZ * DOCLEN * SENTLEN * VOCAB;  // dec_outs
  const size_t OFF2 = OFF1 + (size_t)BSZ * DOCLEN * SRCLEN;    // attn_scores (zeros)
  const size_t OFF3 = OFF2 + (size_t)NB2 * SENTLEN * SRCLEN;   // word_atte_scores

  // workspace carve (256-byte aligned blocks)
  char* ws = (char*)d_ws;
  size_t off = 0;
  auto carve = [&](size_t bytes) -> char* {
    char* p = ws + off;
    off += (bytes + 255) & ~(size_t)255;
    return p;
  };
  bf16_t* wihd_b = (bf16_t*)carve((size_t)GATES * HID * 2);
  bf16_t* whhd_b = (bf16_t*)carve((size_t)GATES * HID * 2);
  bf16_t* wihw_b = (bf16_t*)carve((size_t)GATES * HID * 2);
  bf16_t* whhw_b = (bf16_t*)carve((size_t)GATES * HID * 2);
  bf16_t* aqd_b  = (bf16_t*)carve((size_t)HID * HID * 2);
  bf16_t* aod_b  = (bf16_t*)carve((size_t)HID * 2 * HID * 2);
  bf16_t* aqw_b  = (bf16_t*)carve((size_t)HID * HID * 2);
  bf16_t* aow_b  = (bf16_t*)carve((size_t)HID * 2 * HID * 2);
  bf16_t* fcw_b  = (bf16_t*)carve((size_t)VOCAB * HID * 2);
  bf16_t* enc1_b = (bf16_t*)carve((size_t)BSZ * SRCLEN * HID * 2);
  bf16_t* enc1_t = (bf16_t*)carve((size_t)BSZ * HID * SRCPAD * 2);
  float*  sstate = (float*)carve((size_t)NB2 * HID * 4);
  bf16_t* wouts  = (bf16_t*)carve((size_t)NB2 * SENTLEN * HID * 2);

  // ---- prep: bf16 conversions, encoder transpose, zero attn_scores ----
  const int CT = 256, CB = 2048;
  k_cvt<<<CB, CT, 0, stream>>>(wihd, wihd_b, GATES * HID);
  k_cvt<<<CB, CT, 0, stream>>>(whhd, whhd_b, GATES * HID);
  k_cvt<<<CB, CT, 0, stream>>>(wihw, wihw_b, GATES * HID);
  k_cvt<<<CB, CT, 0, stream>>>(whhw, whhw_b, GATES * HID);
  k_cvt<<<CB, CT, 0, stream>>>(aqd, aqd_b, HID * HID);
  k_cvt<<<CB, CT, 0, stream>>>(aod, aod_b, HID * 2 * HID);
  k_cvt<<<CB, CT, 0, stream>>>(aqw, aqw_b, HID * HID);
  k_cvt<<<CB, CT, 0, stream>>>(aow, aow_b, HID * 2 * HID);
  k_cvt<<<CB, CT, 0, stream>>>(fcw, fcw_b, VOCAB * HID);
  k_cvt<<<CB, CT, 0, stream>>>(enc1, enc1_b, BSZ * SRCLEN * HID);
  k_tr<<<CB, CT, 0, stream>>>(enc1, enc1_t);
  k_zero<<<64, CT, 0, stream>>>(out + OFF1, BSZ * DOCLEN * SRCLEN);

  // ---- document-level scan ----
  k_doc<<<1, 256, 0, stream>>>(enc0, mask, emb, wihd_b, whhd_b, bihd, bhhd,
                               aqd_b, aod_b, sstate, out + OFF3);

  // ---- word-level scan (8 independent WGs, one per shared-encoder batch) ----
  k_word<<<BSZ, 256, 0, stream>>>(tokens, spe, emb, enc1_b, enc1_t, mask,
                                  wihw_b, whhw_b, bihw, bhhw,
                                  aqw_b, aow_b, sstate, wouts, out + OFF2);

  // ---- vocabulary projection (store-bandwidth-bound; bf16 WMMA, f32 out) ----
  dim3 fgrid(VOCAB / 128, (NB2 * SENTLEN) / 64);  // 250 x 80
  k_fc<<<fgrid, 256, 0, stream>>>(wouts, fcw_b, fcb, out);
}
